// AttentionUpscaling_53188874993775
// MI455X (gfx1250) — matrix operand order repaired and checked by reading, not compiled, not run on hardware
//
#include <hip/hip_runtime.h>

typedef __attribute__((ext_vector_type(16))) _Float16 v16h;
typedef __attribute__((ext_vector_type(8)))  float    v8f;

#define B_SZ  4
#define C_SZ  3
#define HH    512
#define HL    256
#define SPAT  8      // HR stride
#define NP    63     // patches per dim
#define L_DIM 3969   // NP*NP
#define N_DIM 768    // C*16*16
#define KPAD  4000   // L padded to multiple of 32
#define KB_N  (KPAD / 32)   // 125 K-blocks
#define TM    128
#define TN    128
#define TK    32
#define A_STR 40     // LDS row stride (halves): 80B rows -> 16B aligned, 20-bank step

// Bilinear 2x upsample, half-pixel centers (align_corners=False):
// fixed 0.75/0.25 weights by output parity, edge-clamped.
__device__ __forceinline__ float bilin2x(const float* __restrict__ img, int y, int x) {
    const int ym = y >> 1, xm = x >> 1;
    int y0, y1, x0, x1; float wy0, wx0;
    if (y & 1) { y0 = ym; y1 = (ym + 1 < HL) ? ym + 1 : HL - 1; wy0 = 0.75f; }
    else       { y0 = (ym > 0) ? ym - 1 : 0; y1 = ym;           wy0 = 0.25f; }
    if (x & 1) { x0 = xm; x1 = (xm + 1 < HL) ? xm + 1 : HL - 1; wx0 = 0.75f; }
    else       { x0 = (xm > 0) ? xm - 1 : 0; x1 = xm;           wx0 = 0.25f; }
    const float wy1 = 1.0f - wy0, wx1 = 1.0f - wx0;
    const float a = img[y0 * HL + x0], b = img[y0 * HL + x1];
    const float c = img[y1 * HL + x0], d = img[y1 * HL + x1];
    return wy0 * (wx0 * a + wx1 * b) + wy1 * (wx0 * c + wx1 * d);
}

// hf stored in WMMA-B-fragment-native tiled layout:
//   hfT[((b*KB_N + k/32)*N_DIM + d)*32 + (k%32)]   (f16)
// so the GEMM loads B fragments straight from global with b128s (no LDS repack).
// K rows [L_DIM, KPAD) are zero -> A garbage in the K tail contributes nothing.
__global__ __launch_bounds__(256) void build_hf(
    const float* __restrict__ x_hr, const float* __restrict__ x_lr_blur,
    _Float16* __restrict__ hfT)
{
    const int idx   = blockIdx.x * blockDim.x + threadIdx.x;
    const int total = B_SZ * KPAD * N_DIM;
    if (idx >= total) return;
    const int kk = idx & 31;
    const int d  = (idx >> 5) % N_DIM;
    const int t  = idx / (32 * N_DIM);
    const int kb = t % KB_N;
    const int b  = t / KB_N;
    const int p  = kb * 32 + kk;           // patch index (GEMM K)
    _Float16 val = (_Float16)0.0f;
    if (p < L_DIM) {
        const int c  = d >> 8;             // d / 256
        const int r  = d & 255;
        const int ky = r >> 4, kx = r & 15;
        const int py = p / NP, px = p % NP;
        const int y  = py * SPAT + ky, x = px * SPAT + kx;
        const float hr = x_hr[(((size_t)b * C_SZ + c) * HH + y) * HH + x];
        const float bl = bilin2x(x_lr_blur + ((size_t)b * C_SZ + c) * HL * HL, y, x);
        val = (_Float16)(hr - bl);
    }
    hfT[idx] = val;
}

// rec[b,l,d] = sum_m attn[b,l,m] * hf[b,m,d]
// A = attn fp32 -> f16 on the fly through LDS (63MB/batch, L2-resident),
// B = hfT f16 fragments direct from global (6MB/batch, L2-resident).
__global__ __launch_bounds__(256) void attn_gemm(
    const float* __restrict__ attn,
    const _Float16* __restrict__ hfT,
    float* __restrict__ rec)
{
    __shared__ alignas(16) _Float16 As[2][TM][A_STR];   // ping-pong, 20KB

    const int tid  = threadIdx.x;
    const int lane = tid & 31;
    const int wave = tid >> 5;
    const int hi   = lane >> 4;    // half-wave 0/1
    const int ln   = lane & 15;

    const int n0 = blockIdx.x * TN;
    const int m0 = blockIdx.y * TM;
    const int b  = blockIdx.z;

    const float* attnB = attn + (size_t)b * L_DIM * L_DIM;

    const int wm = (wave >> 1) * 32;   // 8 waves: 4x2 grid of 32x64 sub-tiles
    const int wn = (wave & 1) * 64;

    const int arow = tid >> 3;         // 0..31  (x4 -> 128 rows)
    const int acol = (tid & 7) * 4;    // 0..28  (4 floats -> 32 cols)

    // Unguarded A staging: clamp the ROW only. For m<=3968, k<=3999:
    // m*3969 + k <= 3968*3969 + 3999 = 3969*3969 + 30, so the last row's loads run up to
    // 31 floats past this batch's attn; K-tail garbage multiplies hf's zero rows -> contributes 0.
    auto stageA = [&](int k0, int buf) {
        #pragma unroll
        for (int i = 0; i < 4; ++i) {
            const int rr = arow + 32 * i;
            const int mm = m0 + rr;
            const int m  = (mm < L_DIM) ? mm : (L_DIM - 1);
            const float* p = attnB + (size_t)m * L_DIM + (k0 + acol);
            const float v0 = p[0], v1 = p[1], v2 = p[2], v3 = p[3];
            if (k0 + TK < KPAD) __builtin_prefetch(p + TK, 0, 0);  // next K tile
            union { _Float16 h[4]; uint2 u; } pk;
            pk.h[0] = (_Float16)v0; pk.h[1] = (_Float16)v1;
            pk.h[2] = (_Float16)v2; pk.h[3] = (_Float16)v3;
            *reinterpret_cast<uint2*>(&As[buf][rr][acol]) = pk.u;
        }
    };

    v8f acc[2][4] = {};

    stageA(0, 0);
    __syncthreads();

    int cur = 0;
    for (int kb = 0; kb < KB_N; ++kb) {
        if (kb + 1 < KB_N) stageA((kb + 1) * TK, cur ^ 1);

        // A frags: ISA 16-bit 16x32 layout -> halves e0..7 = K hi*8+0..7,
        // e8..15 = K 16+hi*8+0..7: two 16B-contiguous runs -> 2x ds_load_b128.
        union { v16h v; uint4 q[2]; } a[2], bf[4];
        #pragma unroll
        for (int i = 0; i < 2; ++i) {
            const _Float16* ap = &As[cur][wm + i * 16 + ln][hi * 8];
            a[i].q[0] = *reinterpret_cast<const uint4*>(ap);
            a[i].q[1] = *reinterpret_cast<const uint4*>(ap + 16);
        }
        // B frags: lane ln = col, halves e0..7 = K hi*8+0..7, e8..15 = K 16+hi*8+0..7
        // in the tiled hfT layout -> 2x global_load_b128 per frag.
        #pragma unroll
        for (int j = 0; j < 4; ++j) {
            const _Float16* bp = hfT
                + (((size_t)b * KB_N + kb) * N_DIM + (n0 + wn + j * 16 + ln)) * 32
                + hi * 8;
            bf[j].q[0] = *reinterpret_cast<const uint4*>(bp);
            bf[j].q[1] = *reinterpret_cast<const uint4*>(bp + 16);
        }

        #pragma unroll
        for (int i = 0; i < 2; ++i)
            #pragma unroll
            for (int j = 0; j < 4; ++j)
                acc[i][j] = __builtin_amdgcn_wmma_f32_16x16x32_f16(
                    false, a[i].v, false, bf[j].v, (short)0, acc[i][j], false, false);

        __syncthreads();   // As[cur] reads done; As[cur^1] writes visible
        cur ^= 1;
    }

    // D layout: VGPR r -> M = r + 8*hi, N = ln
    #pragma unroll
    for (int i = 0; i < 2; ++i) {
        #pragma unroll
        for (int r = 0; r < 8; ++r) {
            const int m = m0 + wm + i * 16 + hi * 8 + r;
            if (m >= L_DIM) continue;
            float* dst = rec + ((size_t)b * L_DIM + m) * N_DIM + n0 + wn + ln;
            #pragma unroll
            for (int j = 0; j < 4; ++j)
                dst[j * 16] = acc[i][j][r];
        }
    }
}

// out = bilin2x(x_lr_inpainted) + fold(rec)/fold(ones).  k=16, s=8 -> each
// pixel covered by <=2x2 patches; integer count is the exact overlap norm.
__global__ __launch_bounds__(256) void fold_add(
    const float* __restrict__ rec, const float* __restrict__ x_lr_inp,
    float* __restrict__ out)
{
    const int idx   = blockIdx.x * blockDim.x + threadIdx.x;
    const int total = B_SZ * C_SZ * HH * HH;
    if (idx >= total) return;
    const int x = idx & (HH - 1);
    const int y = (idx >> 9) & (HH - 1);
    const int c = (idx / (HH * HH)) % C_SZ;
    const int b = idx / (HH * HH * C_SZ);

    const float base = bilin2x(x_lr_inp + ((size_t)b * C_SZ + c) * HL * HL, y, x);

    float sum = 0.0f; int cnt = 0;
    const int pyb = y >> 3, pxb = x >> 3;
    #pragma unroll
    for (int dy = -1; dy <= 0; ++dy) {
        const int py = pyb + dy;
        if (py < 0 || py >= NP) continue;
        const int ky = y - py * SPAT;          // in [0,16) by construction
        #pragma unroll
        for (int dx = -1; dx <= 0; ++dx) {
            const int px = pxb + dx;
            if (px < 0 || px >= NP) continue;
            const int kx = x - px * SPAT;
            sum += rec[((size_t)b * L_DIM + (py * NP + px)) * N_DIM
                       + (c << 8) + ky * 16 + kx];
            ++cnt;
        }
    }
    out[idx] = base + sum / (float)cnt;
}

extern "C" void kernel_launch(void* const* d_in, const int* in_sizes, int n_in,
                              void* d_out, int out_size, void* d_ws, size_t ws_size,
                              hipStream_t stream) {
    const float* x_hr   = (const float*)d_in[0];
    const float* x_lr_i = (const float*)d_in[1];
    const float* attn   = (const float*)d_in[2];
    const float* x_lr_b = (const float*)d_in[3];
    float* out = (float*)d_out;

    // workspace: hfT f16 tiled (24.6 MB) | rec f32 (48.8 MB)
    _Float16* hfT = (_Float16*)d_ws;
    const size_t hf_bytes = (size_t)B_SZ * KPAD * N_DIM * sizeof(_Float16);
    float* rec = (float*)((char*)d_ws + ((hf_bytes + 255) & ~(size_t)255));

    {
        const int n = B_SZ * KPAD * N_DIM;
        build_hf<<<(n + 255) / 256, 256, 0, stream>>>(x_hr, x_lr_b, hfT);
    }
    {
        dim3 grid(N_DIM / TN, (L_DIM + TM - 1) / TM, B_SZ);  // 6 x 32 x 4
        attn_gemm<<<grid, 256, 0, stream>>>(attn, hfT, rec);
    }
    {
        const int n = B_SZ * C_SZ * HH * HH;
        fold_add<<<(n + 255) / 256, 256, 0, stream>>>(rec, x_lr_i, out);
    }
}
